// DPQJointClassLoss_61916248539750
// MI455X (gfx1250) — matrix operand order repaired and checked by reading, not compiled, not run on hardware
//
#include <hip/hip_runtime.h>
#include <hip/hip_bf16.h>
#include <hip/hip_fp16.h>

#define NUM_CLASS   10000
#define FEATURE_DIM 512
#define BATCH       4096
#define N_CHUNKS    (NUM_CLASS / 16)   // 625 class tiles of 16
#define ROW_BLOCKS  (BATCH / 16)       // 256
#define ROWSTRIDE   520                // 512 + 8 halfs pad: 1040B = 260 dwords == 4 mod 64 banks

typedef __attribute__((ext_vector_type(16))) _Float16 v16h;
typedef __attribute__((ext_vector_type(8)))  _Float16 v8h;
typedef __attribute__((ext_vector_type(4)))  _Float16 v4h;
typedef __attribute__((ext_vector_type(8)))  float    v8f;

union AFrag { v16h v; v8h h[2]; };

// ---------------------------------------------------------------- f32 -> f16
__global__ void cvt_f32_f16_kernel(const float* __restrict__ src,
                                   _Float16* __restrict__ dst, int n4) {
    int i = blockIdx.x * blockDim.x + threadIdx.x;
    if (i < n4) {
        float4 f = ((const float4*)src)[i];
        v4h h;
        h.x = (_Float16)f.x; h.y = (_Float16)f.y;
        h.z = (_Float16)f.z; h.w = (_Float16)f.w;
        ((v4h*)dst)[i] = h;
    }
}

// ---------------------------------------------------- fused GEMM + online CE
// grid = ROW_BLOCKS blocks of 256 threads (8 waves). Block stages its 16 rows
// of soft_x and hard_x (f32->f16) into padded LDS once; wave w then handles
// class chunks w, w+8, ... computing a 16x16 logit tile for soft AND hard
// (shared B from global), A fragments from LDS via ds_load_b128. Per-lane
// online logsumexp per row-slot, target logit captured by column compare,
// merged via half-wave butterfly then LDS across waves.
__global__ __launch_bounds__(256)
void gemm_ce_kernel(const float* __restrict__ Xs32,
                    const float* __restrict__ Xh32,
                    const _Float16* __restrict__ W,
                    const int* __restrict__ targets,
                    float* __restrict__ ce_partial) {
    const int m0   = blockIdx.x * 16;
    const int lane = threadIdx.x & 31;
    const int wave = threadIdx.x >> 5;     // 0..7
    const int hw   = lane >> 4;            // half-wave: 0 or 1
    const int lm   = lane & 15;

    // ---- stage A (soft & hard) into LDS, converting f32 -> f16, padded rows
    __shared__ _Float16 lds_a[2 * 16 * ROWSTRIDE];   // soft at 0, hard at 16*ROWSTRIDE
#pragma unroll
    for (int j = 0; j < 8; ++j) {
        int c    = threadIdx.x + j * 256;            // 0..2047 float4 chunks
        int row  = c >> 7;                           // 128 float4 per row
        int col4 = c & 127;
        float4 fs = ((const float4*)(Xs32 + (size_t)(m0 + row) * FEATURE_DIM))[col4];
        float4 fh = ((const float4*)(Xh32 + (size_t)(m0 + row) * FEATURE_DIM))[col4];
        v4h hs, hh;
        hs.x = (_Float16)fs.x; hs.y = (_Float16)fs.y;
        hs.z = (_Float16)fs.z; hs.w = (_Float16)fs.w;
        hh.x = (_Float16)fh.x; hh.y = (_Float16)fh.y;
        hh.z = (_Float16)fh.z; hh.w = (_Float16)fh.w;
        *(v4h*)&lds_a[row * ROWSTRIDE + col4 * 4]                  = hs;
        *(v4h*)&lds_a[16 * ROWSTRIDE + row * ROWSTRIDE + col4 * 4] = hh;
    }
    __syncthreads();

    // A-fragment base for this lane (A: lanes l and l+16 both hold row l%16)
    const _Float16* asbase = &lds_a[lm * ROWSTRIDE];
    const _Float16* ahbase = &lds_a[16 * ROWSTRIDE + lm * ROWSTRIDE];
    const int aoff = hw ? 16 : 0;          // K offset for upper half-wave
    const int ko   = hw ? 16 : 0;          // B K offset for upper half-wave

    // Target class for each of this lane's 8 C/D row-slots (row = r + 8*hw)
    int tgt[8];
#pragma unroll
    for (int r = 0; r < 8; ++r) tgt[r] = targets[m0 + r + 8 * hw];

    float ms[8], ss[8], mh[8], sh[8], tls[8], tlh[8];
#pragma unroll
    for (int r = 0; r < 8; ++r) {
        ms[r] = -1.0e30f; ss[r] = 0.f;
        mh[r] = -1.0e30f; sh[r] = 0.f;
        tls[r] = 0.f;     tlh[r] = 0.f;
    }

    for (int ci = wave; ci < N_CHUNKS; ci += 8) {
        const int n0 = ci * 16;
        const _Float16* wrow = W + (size_t)(n0 + lm) * FEATURE_DIM + ko;
        v8f acc_s = {}; v8f acc_h = {};
#pragma unroll 4
        for (int kc = 0; kc < FEATURE_DIM / 32; ++kc) {
            const int k0 = kc * 32;
            AFrag as, ah;
            as.h[0] = *(const v8h*)(asbase + k0 + aoff);
            as.h[1] = *(const v8h*)(asbase + k0 + 8 + aoff);
            ah.h[0] = *(const v8h*)(ahbase + k0 + aoff);
            ah.h[1] = *(const v8h*)(ahbase + k0 + 8 + aoff);
            v16h b  = *(const v16h*)(wrow + k0);
            acc_s = __builtin_amdgcn_wmma_f32_16x16x32_f16(
                        false, as.v, false, b, (short)0, acc_s, false, false);
            acc_h = __builtin_amdgcn_wmma_f32_16x16x32_f16(
                        false, ah.v, false, b, (short)0, acc_h, false, false);
        }
        // per-lane online logsumexp update; lane's column is n0 + lm
        const int mycol = n0 + lm;
#pragma unroll
        for (int r = 0; r < 8; ++r) {
            float ds = acc_s[r];
            float mn = fmaxf(ms[r], ds);
            ss[r] = ss[r] * __expf(ms[r] - mn) + __expf(ds - mn);
            ms[r] = mn;
            float dh = acc_h[r];
            mn = fmaxf(mh[r], dh);
            sh[r] = sh[r] * __expf(mh[r] - mn) + __expf(dh - mn);
            mh[r] = mn;
            if (tgt[r] == mycol) { tls[r] += ds; tlh[r] += dh; }
        }
    }

    // merge across the 16 lanes of each half-wave (xor masks 1,2,4,8 keep halves separate)
    __shared__ float red[8][16][6];   // [wave][row][Ms,Ss,Mh,Sh,Ts,Th]
#pragma unroll
    for (int r = 0; r < 8; ++r) {
        float Ms = ms[r], Ss = ss[r], Mh = mh[r], Sh = sh[r];
        float Ts = tls[r], Th = tlh[r];
#pragma unroll
        for (int off = 1; off < 16; off <<= 1) {
            float Mo = __shfl_xor(Ms, off, 32);
            float So = __shfl_xor(Ss, off, 32);
            float Mn = fmaxf(Ms, Mo);
            Ss = Ss * __expf(Ms - Mn) + So * __expf(Mo - Mn);
            Ms = Mn;
            Mo = __shfl_xor(Mh, off, 32);
            So = __shfl_xor(Sh, off, 32);
            Mn = fmaxf(Mh, Mo);
            Sh = Sh * __expf(Mh - Mn) + So * __expf(Mo - Mn);
            Mh = Mn;
            Ts += __shfl_xor(Ts, off, 32);
            Th += __shfl_xor(Th, off, 32);
        }
        if (lm == 0) {
            int row = r + 8 * hw;
            red[wave][row][0] = Ms; red[wave][row][1] = Ss;
            red[wave][row][2] = Mh; red[wave][row][3] = Sh;
            red[wave][row][4] = Ts; red[wave][row][5] = Th;
        }
    }
    __syncthreads();

    // merge the 8 waves' partials; 16 threads handle 16 rows
    __shared__ float rowloss[16];
    if (threadIdx.x < 16) {
        int row = threadIdx.x;
        float Ms = -1.0e30f, Ss = 0.f, Mh = -1.0e30f, Sh = 0.f, Ts = 0.f, Th = 0.f;
#pragma unroll
        for (int w = 0; w < 8; ++w) {
            float Mo = red[w][row][0], So = red[w][row][1];
            float Mn = fmaxf(Ms, Mo);
            Ss = Ss * __expf(Ms - Mn) + So * __expf(Mo - Mn);
            Ms = Mn;
            Mo = red[w][row][2]; So = red[w][row][3];
            Mn = fmaxf(Mh, Mo);
            Sh = Sh * __expf(Mh - Mn) + So * __expf(Mo - Mn);
            Mh = Mn;
            Ts += red[w][row][4];
            Th += red[w][row][5];
        }
        // -log_softmax at target = logsumexp - target_logit
        rowloss[row] = (Ms + __logf(Ss) - Ts) + (Mh + __logf(Sh) - Th);
    }
    __syncthreads();
    if (threadIdx.x == 0) {
        float s = 0.f;
#pragma unroll
        for (int i = 0; i < 16; ++i) s += rowloss[i];
        ce_partial[blockIdx.x] = s;
    }
}

// -------------------------------------------------- quantization loss kernel
__global__ __launch_bounds__(256)
void quant_kernel(const float* __restrict__ Xs, const float* __restrict__ Xh,
                  const int* __restrict__ targets,
                  const float* __restrict__ centers,
                  float* __restrict__ q_partial) {
    const int row0 = blockIdx.x * 16;
    float acc = 0.f;
    for (int e = threadIdx.x; e < 16 * FEATURE_DIM; e += 256) {
        int r = e >> 9;           // FEATURE_DIM = 512
        int d = e & 511;
        int row = row0 + r;
        int t = targets[row];
        float c = centers[(size_t)t * FEATURE_DIM + d];
        float a = Xs[(size_t)row * FEATURE_DIM + d] - c;
        float b = Xh[(size_t)row * FEATURE_DIM + d] - c;
        acc += a * a + b * b;
    }
    __shared__ float sm[256];
    sm[threadIdx.x] = acc;
    __syncthreads();
    for (int s = 128; s > 0; s >>= 1) {
        if (threadIdx.x < (unsigned)s) sm[threadIdx.x] += sm[threadIdx.x + s];
        __syncthreads();
    }
    if (threadIdx.x == 0) q_partial[blockIdx.x] = sm[0];
}

// -------------------------------------------------------------- final reduce
__global__ __launch_bounds__(256)
void final_kernel(const float* __restrict__ ce_partial,
                  const float* __restrict__ q_partial,
                  float* __restrict__ out) {
    __shared__ float s1[256], s2[256];
    s1[threadIdx.x] = ce_partial[threadIdx.x];
    s2[threadIdx.x] = q_partial[threadIdx.x];
    __syncthreads();
    for (int s = 128; s > 0; s >>= 1) {
        if (threadIdx.x < (unsigned)s) {
            s1[threadIdx.x] += s1[threadIdx.x + s];
            s2[threadIdx.x] += s2[threadIdx.x + s];
        }
        __syncthreads();
    }
    if (threadIdx.x == 0) {
        // loss = mean CE(soft) + mean CE(hard) + PARAM * 0.5 * quant_sum / B
        out[0] = s1[0] / (float)BATCH + 0.25f * s2[0] / (float)BATCH;
    }
}

extern "C" void kernel_launch(void* const* d_in, const int* in_sizes, int n_in,
                              void* d_out, int out_size, void* d_ws, size_t ws_size,
                              hipStream_t stream) {
    const float* soft    = (const float*)d_in[0];
    const float* hard    = (const float*)d_in[1];
    const int*   targets = (const int*)d_in[2];
    const float* centers = (const float*)d_in[3];
    const float* weight  = (const float*)d_in[4];

    _Float16* w16  = (_Float16*)d_ws;
    float* ce_part = (float*)(w16 + (size_t)NUM_CLASS * FEATURE_DIM);
    float* q_part  = ce_part + ROW_BLOCKS;

    const int nw4 = NUM_CLASS * FEATURE_DIM / 4;
    cvt_f32_f16_kernel<<<(nw4 + 255) / 256, 256, 0, stream>>>(weight, w16, nw4);

    gemm_ce_kernel<<<ROW_BLOCKS, 256, 0, stream>>>(soft, hard, w16, targets, ce_part);
    quant_kernel<<<ROW_BLOCKS, 256, 0, stream>>>(soft, hard, targets, centers, q_part);
    final_kernel<<<1, 256, 0, stream>>>(ce_part, q_part, (float*)d_out);
}
